// FlaxFalconAttention_51032801411503
// MI455X (gfx1250) — matrix operand order repaired and checked by reading, not compiled, not run on hardware
//
#include <hip/hip_runtime.h>

// ---------------------------------------------------------------------------
// CDNA5 (gfx1250) Falcon attention: bf16 WMMA GEMMs + flash attention.
// b128 LDS fragment loads, global prefetch, optional TDM A-tile staging.
// ---------------------------------------------------------------------------

typedef __attribute__((ext_vector_type(16))) __bf16        v16bf;
typedef __attribute__((ext_vector_type(8)))  float         v8f;
typedef __attribute__((ext_vector_type(4)))  unsigned int  u32x4;
typedef __attribute__((ext_vector_type(8)))  int           i32x8;
typedef __attribute__((ext_vector_type(4)))  int           i32x4;

#define HEADS    71
#define HEAD_DIM 64
#define HIDDEN   4544
#define QKV_W    13632      // 3 * HIDDEN
#define SEQ      1024
#define BATCH    2
#define TOKENS   2048       // BATCH * SEQ

#if __has_builtin(__builtin_amdgcn_tensor_load_to_lds)
#define HAVE_TDM 1
#endif

__device__ __forceinline__ unsigned short f2bf(float f) {
  unsigned int u = __builtin_bit_cast(unsigned int, f);
  u += 0x7FFFu + ((u >> 16) & 1u);            // round to nearest even
  return (unsigned short)(u >> 16);
}
__device__ __forceinline__ unsigned int pack2bf(float a, float b) {
  return (unsigned int)f2bf(a) | ((unsigned int)f2bf(b) << 16);
}

union FragBF { u32x4 q[2]; unsigned int u[8]; v16bf v; };

// ---------------------------------------------------------------------------
// fp32 -> bf16 conversion (grid-stride)
// ---------------------------------------------------------------------------
__global__ void cvt_bf16_kernel(const float* __restrict__ src,
                                unsigned short* __restrict__ dst, long long n) {
  long long i = (long long)blockIdx.x * blockDim.x + threadIdx.x;
  long long stride = (long long)gridDim.x * blockDim.x;
  for (; i < n; i += stride) dst[i] = f2bf(src[i]);
}

// ---------------------------------------------------------------------------
// bf16 GEMM: C[M,N] = A[M,K] * B[K,N], f32 accumulate.
// Block tile 256x64, K-step 32, 256 threads = 8 waves.
// Wave owns 2 row-tiles x 4 col-tiles -> 8 WMMA per K-step, B frags reused 2x.
// LDS rows padded to 80B so every fragment is 2x ds_load_b128.
// A tile optionally staged by the Tensor Data Mover (pad_enable gives the 80B
// row pitch in LDS); fallback is manual staging.
// ---------------------------------------------------------------------------
#define BM 256
#define BN 64
#define BK 32
#define LDK 40              // 32 bf16 + 8 pad -> 80B row pitch (16B aligned)

__global__ __launch_bounds__(256)
void gemm_bf16_kernel(const unsigned short* __restrict__ A,
                      const unsigned short* __restrict__ B,
                      float* __restrict__ C, int M, int N, int K) {
  __shared__ __attribute__((aligned(16))) unsigned short sA[BM][LDK]; // [m][k]
  __shared__ __attribute__((aligned(16))) unsigned short sB[BN][LDK]; // [n][k]

  const int tid   = threadIdx.x;
  const int wave  = tid >> 5;
  const int lane  = tid & 31;
  const int ln    = lane & 15;
  const int half  = lane >> 4;
  const int m0    = blockIdx.y * BM;
  const int n0    = blockIdx.x * BN;
  const int rbase = wave * 32;

  v8f acc[2][4];
#pragma unroll
  for (int rt = 0; rt < 2; ++rt)
#pragma unroll
    for (int i = 0; i < 4; ++i)
      acc[rt][i] = (v8f){0.f,0.f,0.f,0.f,0.f,0.f,0.f,0.f};

  for (int k0 = 0; k0 < K; k0 += BK) {
    __syncthreads();

#ifdef HAVE_TDM
    // ---- A tile via Tensor Data Mover: 2D tile BKxBM, stride K, data 2B,
    //      LDS padding 16B every 64B -> 80B row pitch (matches LDK).
    if (wave == 0) {
      unsigned lds_a = (unsigned)(size_t)(&sA[0][0]);   // low 32b = LDS offset
      unsigned long long ga = (unsigned long long)(size_t)A +
                              ((unsigned long long)m0 * (unsigned)K + (unsigned)k0) * 2ull;
      u32x4 g0;
      g0[0] = 1u;                                        // count=1, no gather
      g0[1] = lds_a;                                     // lds_addr
      g0[2] = (unsigned)(ga & 0xFFFFFFFFu);              // global_addr[31:0]
      g0[3] = (unsigned)((ga >> 32) & 0x01FFFFFFu) | (2u << 30); // [56:32]|type=2
      i32x8 g1;
      g1[0] = (int)((1u << 16)  |                        // data_size = 2B
                    (1u << 20)  |                        // pad_enable
                    (3u << 22)  |                        // pad_interval: 16 DW (64B)
                    (3u << 25));                         // pad_amount: 4 DW (16B)
      g1[1] = (int)(((unsigned)K & 0xFFFFu) << 16);      // tensor_dim0 lo16
      g1[2] = (int)((((unsigned)K >> 16) & 0xFFFFu) |
                    (((unsigned)M & 0xFFFFu) << 16));    // dim0 hi | dim1 lo
      g1[3] = (int)((((unsigned)M >> 16) & 0xFFFFu) |
                    ((unsigned)BK << 16));               // dim1 hi | tile_dim0
      g1[4] = (int)BM;                                   // tile_dim1 (tile_dim2=0)
      g1[5] = (int)K;                                    // tensor_dim0_stride lo32
      g1[6] = 0;
      g1[7] = 0;
      i32x4 gz = {0, 0, 0, 0};
#if defined(__clang_major__) && (__clang_major__ >= 23)
      i32x8 gz8 = {0, 0, 0, 0, 0, 0, 0, 0};
      __builtin_amdgcn_tensor_load_to_lds(g0, g1, gz, gz, gz8, 0);
#else
      __builtin_amdgcn_tensor_load_to_lds(g0, g1, gz, gz, 0);
#endif
    }
#else
    // ---- Manual A tile: BMxBK bf16 as 32-bit pairs (4096 pairs / 256 thr)
    for (int i = tid; i < BM * (BK / 2); i += 256) {
      int row = i >> 4;                 // 16 pairs per row
      int kc  = (i & 15) << 1;
      unsigned int val =
          *(const unsigned int*)(A + (long long)(m0 + row) * K + k0 + kc);
      *(unsigned int*)&sA[row][kc] = val;
      if (k0 + BK < K)
        __builtin_prefetch(A + (long long)(m0 + row) * K + k0 + BK + kc, 0, 1);
    }
#endif

    // ---- B tile: BKxBN -> transposed [n][k], prefetch next K-tile
    for (int i = tid; i < BK * BN; i += 256) {
      int kk = i >> 6;                  // / BN
      int n  = i & 63;
      sB[n][kk] = B[(long long)(k0 + kk) * N + n0 + n];
      if (k0 + BK < K)
        __builtin_prefetch(B + (long long)(k0 + BK + kk) * N + n0 + n, 0, 1);
    }

#ifdef HAVE_TDM
    if (wave == 0) {
#if __has_builtin(__builtin_amdgcn_s_wait_tensorcnt)
      __builtin_amdgcn_s_wait_tensorcnt(0);
#else
      asm volatile("s_wait_tensorcnt 0x0" ::: "memory");
#endif
    }
#endif
    __syncthreads();

    // ---- A fragments (16-bit 16x32 layout): lane row = ln,
    //      elem i -> k = (i<8 ? i : i+8) + half*8  => two contiguous 16B runs
    FragBF a[2];
#pragma unroll
    for (int rt = 0; rt < 2; ++rt) {
      const int arow = rbase + rt * 16 + ln;
      a[rt].q[0] = *(const u32x4*)&sA[arow][half * 8];
      a[rt].q[1] = *(const u32x4*)&sA[arow][half * 8 + 16];
    }
#pragma unroll
    for (int nb = 0; nb < 4; ++nb) {
      // B fragment: lane col = ln, elem i -> k = (i<8 ? i : i+8) + half*8
      FragBF bfr;
      const int col = nb * 16 + ln;
      bfr.q[0] = *(const u32x4*)&sB[col][half * 8];
      bfr.q[1] = *(const u32x4*)&sB[col][half * 8 + 16];
#pragma unroll
      for (int rt = 0; rt < 2; ++rt)
        acc[rt][nb] = __builtin_amdgcn_wmma_f32_16x16x32_bf16(
            false, a[rt].v, false, bfr.v, (short)0, acc[rt][nb], false, false);
    }
  }

  // C/D layout: VGPR r -> row = r + 8*half, col = ln
#pragma unroll
  for (int rt = 0; rt < 2; ++rt)
#pragma unroll
    for (int nb = 0; nb < 4; ++nb) {
      const int col = n0 + nb * 16 + ln;
#pragma unroll
      for (int r = 0; r < 8; ++r) {
        const int row = m0 + rbase + rt * 16 + r + half * 8;
        C[(long long)row * N + col] = acc[rt][nb][r];
      }
    }
}

// ---------------------------------------------------------------------------
// RoPE on q,k (in-place, fp32 QKV buffer [token][13632])
// ---------------------------------------------------------------------------
__global__ void rope_kernel(float* __restrict__ qkv) {
  const int total = TOKENS * HEADS * (HEAD_DIM / 2);
  int idx = blockIdx.x * blockDim.x + threadIdx.x;
  if (idx >= total) return;
  int p     = idx & 31;
  int hh    = (idx >> 5) % HEADS;
  int token = idx / (HEADS * 32);
  int pos   = token % SEQ;
  float freq = __powf(10000.0f, -(float)p / 32.0f);   // theta^(-2p/64)
  float sn, cs;
  __sincosf((float)pos * freq, &sn, &cs);
  float* q = qkv + (long long)token * QKV_W + hh * HEAD_DIM + 2 * p;
  float* k = q + HIDDEN;
  float qr = q[0], qi = q[1];
  q[0] = qr * cs - qi * sn;
  q[1] = qr * sn + qi * cs;
  float kr = k[0], ki = k[1];
  k[0] = kr * cs - ki * sn;
  k[1] = kr * sn + ki * cs;
}

// ---------------------------------------------------------------------------
// Flash attention: one block = (batch*head, 64-query tile), 4 waves x 16 rows.
// S = Q K^T (WMMA bf16), online softmax, O += P V (WMMA bf16).
// All LDS fragment reads are 2x ds_load_b128 (144B row pitch).
// ---------------------------------------------------------------------------
#define AT_THREADS 128
#define KB 64
#define LDH  72   // 64 bf16 + 8 pad -> 144B pitch (16B aligned)
#define LDKY 72

__global__ __launch_bounds__(AT_THREADS)
void attention_kernel(const float* __restrict__ qkv,
                      const float* __restrict__ mask,
                      unsigned short* __restrict__ attn_bf) {
  __shared__ __attribute__((aligned(16))) unsigned short sK[KB][LDH];         // [key][dim]
  __shared__ __attribute__((aligned(16))) unsigned short sVt[HEAD_DIM][LDKY]; // [dim][key]
  __shared__ __attribute__((aligned(16))) unsigned short sP[4][16][LDKY];     // [row][key]

  const int bh   = blockIdx.x;          // 0..141
  const int b    = bh / HEADS;
  const int h    = bh % HEADS;
  const int qt   = blockIdx.y;          // 0..15
  const int tid  = threadIdx.x;
  const int wave = tid >> 5;
  const int lane = tid & 31;
  const int ln   = lane & 15;
  const int half = lane >> 4;
  const int qbase_s = qt * 64;

  // Q fragments for this wave's 16 rows, held across the whole K loop.
  FragBF qf[2];
  {
    const int qrow_s = qbase_s + wave * 16 + ln;   // A-layout: lane row = ln
    const float* qp =
        qkv + (long long)(b * SEQ + qrow_s) * QKV_W + h * HEAD_DIM;
#pragma unroll
    for (int c = 0; c < 2; ++c) {
#pragma unroll
      for (int j = 0; j < 4; ++j) {
        int d0 = c * 32 + half * 8 + 2 * j;
        qf[c].u[j]     = pack2bf(qp[d0], qp[d0 + 1]);
        qf[c].u[j + 4] = pack2bf(qp[d0 + 16], qp[d0 + 17]);
      }
    }
  }

  const float scale = 0.125f;           // 1/sqrt(64)
  float rowmax[8], rowsum[8];
  v8f o[4];
#pragma unroll
  for (int r = 0; r < 8; ++r) { rowmax[r] = -3.0e38f; rowsum[r] = 0.0f; }
#pragma unroll
  for (int i = 0; i < 4; ++i) o[i] = (v8f){0.f,0.f,0.f,0.f,0.f,0.f,0.f,0.f};

  const float* kbase =
      qkv + (long long)b * SEQ * QKV_W + HIDDEN + h * HEAD_DIM;
  const float* vbase =
      qkv + (long long)b * SEQ * QKV_W + 2 * HIDDEN + h * HEAD_DIM;

  for (int kb = 0; kb <= qt; ++kb) {    // causal: skip fully-masked key blocks
    __syncthreads();
    // Stage K [key][dim] and V^T [dim][key] as bf16.
    for (int i = tid; i < KB * (HEAD_DIM / 2); i += AT_THREADS) {
      int key = i >> 5;                 // 32 pairs per key
      int d   = (i & 31) << 1;
      const float* kp = kbase + (long long)(kb * KB + key) * QKV_W + d;
      *(unsigned int*)&sK[key][d] = pack2bf(kp[0], kp[1]);
      const float* vp = vbase + (long long)(kb * KB + key) * QKV_W + d;
      sVt[d][key]     = f2bf(vp[0]);
      sVt[d + 1][key] = f2bf(vp[1]);
    }
    __syncthreads();

    // S = Q K^T: 16 rows x 64 keys per wave (8 WMMA).
    v8f s[4];
#pragma unroll
    for (int nb = 0; nb < 4; ++nb) {
      s[nb] = (v8f){0.f,0.f,0.f,0.f,0.f,0.f,0.f,0.f};
      const int col = nb * 16 + ln;     // key within block
#pragma unroll
      for (int c = 0; c < 2; ++c) {
        FragBF bfr;                     // elem i -> dim = c*32 + (i<8?i:i+8) + half*8
        bfr.q[0] = *(const u32x4*)&sK[col][c * 32 + half * 8];
        bfr.q[1] = *(const u32x4*)&sK[col][c * 32 + half * 8 + 16];
        s[nb] = __builtin_amdgcn_wmma_f32_16x16x32_bf16(
            false, qf[c].v, false, bfr.v, (short)0, s[nb], false, false);
      }
    }

    // Scale + additive causal mask (reference semantics).
#pragma unroll
    for (int nb = 0; nb < 4; ++nb) {
      const int key_s = kb * KB + nb * 16 + ln;
#pragma unroll
      for (int r = 0; r < 8; ++r) {
        const int qrow_s = qbase_s + wave * 16 + r + half * 8;
        s[nb][r] = s[nb][r] * scale + mask[(long long)qrow_s * SEQ + key_s];
      }
    }

    // Online softmax: rows live in (r, half); columns across 16 lanes x 4 nb.
    float corr[8];
#pragma unroll
    for (int r = 0; r < 8; ++r) {
      float m = fmaxf(fmaxf(s[0][r], s[1][r]), fmaxf(s[2][r], s[3][r]));
#pragma unroll
      for (int off = 1; off < 16; off <<= 1)
        m = fmaxf(m, __shfl_xor(m, off, 32));
      float nm = fmaxf(rowmax[r], m);
      corr[r] = __expf(rowmax[r] - nm);
      rowmax[r] = nm;
    }
#pragma unroll
    for (int nb = 0; nb < 4; ++nb)
#pragma unroll
      for (int r = 0; r < 8; ++r)
        s[nb][r] = __expf(s[nb][r] - rowmax[r]);
#pragma unroll
    for (int r = 0; r < 8; ++r) {
      float sum = s[0][r] + s[1][r] + s[2][r] + s[3][r];
#pragma unroll
      for (int off = 1; off < 16; off <<= 1)
        sum += __shfl_xor(sum, off, 32);
      rowsum[r] = rowsum[r] * corr[r] + sum;
#pragma unroll
      for (int i = 0; i < 4; ++i) o[i][r] *= corr[r];
    }

    // P -> LDS (bf16), C-layout row = r + 8*half, col = nb*16+ln.
#pragma unroll
    for (int nb = 0; nb < 4; ++nb)
#pragma unroll
      for (int r = 0; r < 8; ++r)
        sP[wave][r + 8 * half][nb * 16 + ln] = f2bf(s[nb][r]);
    __syncthreads();

    // O += P * V: A = P (16 x 64 keys), B = V^T (8 WMMA).
#pragma unroll
    for (int pc = 0; pc < 2; ++pc) {
      FragBF pf;                        // elem i -> key = pc*32 + (i<8?i:i+8) + half*8
      pf.q[0] = *(const u32x4*)&sP[wave][ln][pc * 32 + half * 8];
      pf.q[1] = *(const u32x4*)&sP[wave][ln][pc * 32 + half * 8 + 16];
#pragma unroll
      for (int dg = 0; dg < 4; ++dg) {
        FragBF vf;                      // elem i -> key = pc*32 + (i<8?i:i+8) + half*8
        const int dcol = dg * 16 + ln;
        vf.q[0] = *(const u32x4*)&sVt[dcol][pc * 32 + half * 8];
        vf.q[1] = *(const u32x4*)&sVt[dcol][pc * 32 + half * 8 + 16];
        o[dg] = __builtin_amdgcn_wmma_f32_16x16x32_bf16(
            false, pf.v, false, vf.v, (short)0, o[dg], false, false);
      }
    }
  }

  // Normalize and store as bf16 in [token][h*64+dim] for the out-projection.
#pragma unroll
  for (int r = 0; r < 8; ++r) {
    const int qrow_s = qbase_s + wave * 16 + r + half * 8;
    const long long tok = (long long)(b * SEQ + qrow_s);
    const float inv = 1.0f / rowsum[r];
#pragma unroll
    for (int dg = 0; dg < 4; ++dg)
      attn_bf[tok * HIDDEN + h * HEAD_DIM + dg * 16 + ln] =
          f2bf(o[dg][r] * inv);
  }
}

// ---------------------------------------------------------------------------
// Launch
// ---------------------------------------------------------------------------
static size_t align256(size_t x) { return (x + 255) & ~(size_t)255; }

extern "C" void kernel_launch(void* const* d_in, const int* in_sizes, int n_in,
                              void* d_out, int out_size, void* d_ws, size_t ws_size,
                              hipStream_t stream) {
  (void)in_sizes; (void)n_in; (void)out_size; (void)ws_size;
  const float* hidden = (const float*)d_in[0];   // [2,1024,4544]
  const float* mask   = (const float*)d_in[1];   // [1,1,1024,1024]
  const float* wqkv   = (const float*)d_in[2];   // [4544,13632]
  const float* wo     = (const float*)d_in[3];   // [4544,4544]
  float* out = (float*)d_out;                    // [2,1024,4544] fp32

  char* ws = (char*)d_ws;
  size_t off = 0;
  unsigned short* hbf     = (unsigned short*)(ws + off);
  off = align256(off + (size_t)TOKENS * HIDDEN * 2);
  unsigned short* wqkv_bf = (unsigned short*)(ws + off);
  off = align256(off + (size_t)HIDDEN * QKV_W * 2);
  unsigned short* wo_bf   = (unsigned short*)(ws + off);
  off = align256(off + (size_t)HIDDEN * HIDDEN * 2);
  float* qkv              = (float*)(ws + off);
  off = align256(off + (size_t)TOKENS * QKV_W * 4);
  unsigned short* attn_bf = (unsigned short*)(ws + off);
  off = align256(off + (size_t)TOKENS * HIDDEN * 2);

  cvt_bf16_kernel<<<1024, 256, 0, stream>>>(hidden, hbf,
                                            (long long)TOKENS * HIDDEN);
  cvt_bf16_kernel<<<4096, 256, 0, stream>>>(wqkv, wqkv_bf,
                                            (long long)HIDDEN * QKV_W);
  cvt_bf16_kernel<<<2048, 256, 0, stream>>>(wo, wo_bf,
                                            (long long)HIDDEN * HIDDEN);

  // QKV projection: [2048,4544] x [4544,13632]
  gemm_bf16_kernel<<<dim3(QKV_W / BN, TOKENS / BM), 256, 0, stream>>>(
      hbf, wqkv_bf, qkv, TOKENS, QKV_W, HIDDEN);

  const int rope_total = TOKENS * HEADS * (HEAD_DIM / 2);
  rope_kernel<<<(rope_total + 255) / 256, 256, 0, stream>>>(qkv);

  // Flash attention: grid (b*h, seq/64)
  attention_kernel<<<dim3(BATCH * HEADS, SEQ / 64), AT_THREADS, 0, stream>>>(
      qkv, mask, attn_bf);

  // Output projection: [2048,4544] x [4544,4544]
  gemm_bf16_kernel<<<dim3(HIDDEN / BN, TOKENS / BM), 256, 0, stream>>>(
      attn_bf, wo_bf, out, TOKENS, HIDDEN, HIDDEN);
}
